// SelfAttention_1434519076891
// MI455X (gfx1250) — matrix operand order repaired and checked by reading, not compiled, not run on hardware
//
#include <hip/hip_runtime.h>
#include <hip/hip_bf16.h>
#include <hip/hip_fp16.h>

// Problem dims (match the reference)
#define BATCH 4
#define SEQ   2048
#define DDIM  1024
#define HDIM  1024
#define MTOT  (BATCH * SEQ)   // 8192 rows for the fused QKV projection
#define BK    64              // K-depth staged per barrier (2 WMMA K-steps)

// ---- CDNA5 WMMA types ----
typedef __attribute__((ext_vector_type(16))) __bf16 bf16x16;   // A/B frag: 16 bf16 = 8 VGPRs
typedef __attribute__((ext_vector_type(8)))  float  f32x8;     // C/D frag: 8 f32  = 8 VGPRs
typedef __attribute__((ext_vector_type(4)))  float  f32x4;
typedef __attribute__((ext_vector_type(4)))  unsigned int u32x4;
typedef __attribute__((ext_vector_type(8)))  unsigned int u32x8;

// ---- fp32 -> bf16 (round-to-nearest-even) ----
__device__ __forceinline__ unsigned int pack2bf(float a, float b) {
  unsigned int ua = __float_as_uint(a);
  unsigned int ub = __float_as_uint(b);
  ua = (ua + 0x7FFFu + ((ua >> 16) & 1u)) >> 16;
  ub = (ub + 0x7FFFu + ((ub >> 16) & 1u)) >> 16;
  return (ua & 0xFFFFu) | (ub << 16);
}
__device__ __forceinline__ unsigned short f2bf(float a) {
  unsigned int ua = __float_as_uint(a);
  return (unsigned short)((ua + 0x7FFFu + ((ua >> 16) & 1u)) >> 16);
}

// ---- Stage a 128xBK tile into LDS as bf16 (row-major, row stride BK) ----
// 256 threads: thread t handles row t>>1, columns (t&1)*32 .. +31.
__device__ __forceinline__ void stage_f32_tile(unsigned short* lds,
                                               const float* __restrict__ g,
                                               size_t ld) {
  const int t   = threadIdx.x;
  const int row = t >> 1;
  const int cg  = (t & 1) << 5;                 // 0 or 32 columns
  const f32x4* ga = reinterpret_cast<const f32x4*>(g + (size_t)row * ld + cg);
  u32x4* dst = reinterpret_cast<u32x4*>(lds + row * BK + cg);
#pragma unroll
  for (int i = 0; i < 4; ++i) {
    f32x4 fa = ga[2 * i];
    f32x4 fb = ga[2 * i + 1];
    u32x4 o;
    o.x = pack2bf(fa.x, fa.y);
    o.y = pack2bf(fa.z, fa.w);
    o.z = pack2bf(fb.x, fb.y);
    o.w = pack2bf(fb.z, fb.w);
    dst[i] = o;
  }
}

__device__ __forceinline__ void stage_bf16_tile(unsigned short* lds,
                                                const unsigned short* __restrict__ g,
                                                size_t ld) {
  const int t   = threadIdx.x;
  const int row = t >> 1;
  const int cg  = (t & 1) << 5;
  const u32x4* ga = reinterpret_cast<const u32x4*>(g + (size_t)row * ld + cg);
  u32x4* dst = reinterpret_cast<u32x4*>(lds + row * BK + cg);
#pragma unroll
  for (int i = 0; i < 4; ++i) dst[i] = ga[i];
}

// ---- Fragment loaders from LDS (contiguous-K order k = 16*(l>>4) + s, same for A and B, wave32) ----
// c = which 32-deep K sub-chunk (0 or 1) within the staged BK=64 tile.
// A 16x32: lane l holds row m=l&15; VGPR0-7 = K (l>>4)*16 .. +15 (16 contiguous K).
__device__ __forceinline__ bf16x16 load_a_frag(const unsigned short* lds, int m0, int c) {
  const int lane = threadIdx.x & 31;
  const int m = lane & 15, h = lane >> 4;
  const unsigned short* base = lds + (m0 + m) * BK + c * 32;
  const u32x4* p0 = reinterpret_cast<const u32x4*>(base + h * 16);
  const u32x4* p1 = reinterpret_cast<const u32x4*>(base + h * 16 + 8);
  u32x4 lo = *p0, hi = *p1;
  u32x8 t = __builtin_shufflevector(lo, hi, 0, 1, 2, 3, 4, 5, 6, 7);
  return __builtin_bit_cast(bf16x16, t);
}
// B 32x16: lane l holds col n=l&15; VGPR v = K (l>>4)*16 + 2v,2v+1 -> 16 contiguous K.
__device__ __forceinline__ bf16x16 load_b_frag(const unsigned short* lds, int n0, int c) {
  const int lane = threadIdx.x & 31;
  const int n = lane & 15, h = lane >> 4;
  const u32x8* p =
      reinterpret_cast<const u32x8*>(lds + (n0 + n) * BK + c * 32 + h * 16);
  u32x8 t = *p;
  return __builtin_bit_cast(bf16x16, t);
}

__device__ __forceinline__ f32x8 wmma_bf16(bf16x16 a, bf16x16 b, f32x8 c) {
  return __builtin_amdgcn_wmma_f32_16x16x32_bf16(false, a, false, b, (short)0, c,
                                                 false, false);
}

// One 16(M) x 128(N) x 64(K) wave-level MMA step from a staged LDS tile pair.
// B-fragment loads are software-pipelined one ahead of the consuming WMMA so
// the matrix pipe is not stalled on ds_load latency.
__device__ __forceinline__ void mma_tile(const unsigned short* as,
                                         const unsigned short* bs, int m0,
                                         f32x8 acc[8]) {
  bf16x16 a0 = load_a_frag(as, m0, 0);
  bf16x16 a1 = load_a_frag(as, m0, 1);
  bf16x16 b0 = load_b_frag(bs, 0, 0);
#pragma unroll
  for (int j = 0; j < 8; ++j) {
    bf16x16 b1 = load_b_frag(bs, j * 16, 1);
    acc[j] = wmma_bf16(a0, b0, acc[j]);
    if (j < 7) b0 = load_b_frag(bs, (j + 1) * 16, 0);
    acc[j] = wmma_bf16(a1, b1, acc[j]);
  }
}

// =====================================================================
// Kernel 1: fused QKV projection.  out[m,h] = sum_d x[m,d] * W[h,d]
// grid = (MTOT/128, HDIM/128, 3); z selects (x1,Wq)->Q, (x2,Wk)->K, (x3,Wv)->Vt
// V is stored TRANSPOSED (Vt[H][MTOT]) so the PV GEMM reads contiguous rows
// and its epilogue becomes b128 stores here.
// =====================================================================
__global__ __launch_bounds__(256) void qkv_proj_kernel(
    const float* __restrict__ x1, const float* __restrict__ x2,
    const float* __restrict__ x3, const float* __restrict__ Wq,
    const float* __restrict__ Wk, const float* __restrict__ Wv,
    unsigned short* __restrict__ qb, unsigned short* __restrict__ kb,
    unsigned short* __restrict__ vtb) {
  __shared__ unsigned short as[2][128 * BK];
  __shared__ unsigned short bs[2][128 * BK];
  const int z = blockIdx.z;
  const float* X = (z == 0) ? x1 : (z == 1) ? x2 : x3;
  const float* W = (z == 0) ? Wq : (z == 1) ? Wk : Wv;
  const int mBase = blockIdx.x * 128;
  const int nBase = blockIdx.y * 128;
  const int w = threadIdx.x >> 5;
  const int lane = threadIdx.x & 31;

  f32x8 acc[8] = {};
  const int NIT = DDIM / BK;  // 16
  stage_f32_tile(as[0], X + (size_t)mBase * DDIM, DDIM);
  stage_f32_tile(bs[0], W + (size_t)nBase * DDIM, DDIM);
  __syncthreads();
  for (int it = 0; it < NIT; ++it) {
    const int cur = it & 1;
    if (it + 1 < NIT) {  // stage next tile while this one is computed
      const int kk = (it + 1) * BK;
      stage_f32_tile(as[1 ^ cur], X + (size_t)mBase * DDIM + kk, DDIM);
      stage_f32_tile(bs[1 ^ cur], W + (size_t)nBase * DDIM + kk, DDIM);
    }
    if (it + 2 < NIT) {  // gfx1250 global_prefetch_b8, two tiles ahead
      __builtin_prefetch(
          X + (size_t)(mBase + (threadIdx.x >> 1)) * DDIM + (it + 2) * BK, 0, 3);
      __builtin_prefetch(
          W + (size_t)(nBase + (threadIdx.x >> 1)) * DDIM + (it + 2) * BK, 0, 3);
    }
    mma_tile(as[cur], bs[cur], w * 16, acc);
    __syncthreads();
  }

  // C/D layout: VGPR r, lane l -> M = (l>>4)*8 + r, N = l&15
  const int n = lane & 15, h = lane >> 4;
  if (z == 2) {
    // Transposed store: per lane the 8 accumulator rows are CONSECUTIVE in
    // Vt's fast dimension -> pack to one b128 store per tile.
#pragma unroll
    for (int j = 0; j < 8; ++j) {
      u32x4 pk;
      pk.x = pack2bf(acc[j][0], acc[j][1]);
      pk.y = pack2bf(acc[j][2], acc[j][3]);
      pk.z = pack2bf(acc[j][4], acc[j][5]);
      pk.w = pack2bf(acc[j][6], acc[j][7]);
      const int col = nBase + j * 16 + n;
      *reinterpret_cast<u32x4*>(vtb + (size_t)col * MTOT + mBase + w * 16 +
                                h * 8) = pk;
    }
  } else {
    unsigned short* dstb = (z == 0) ? qb : kb;
#pragma unroll
    for (int j = 0; j < 8; ++j) {
#pragma unroll
      for (int r = 0; r < 8; ++r) {
        const int mrow = mBase + w * 16 + h * 8 + r;
        const int col = nBase + j * 16 + n;
        dstb[(size_t)mrow * HDIM + col] = f2bf(acc[j][r]);
      }
    }
  }
}

// =====================================================================
// Kernel 2: scores[b,q,k] = (1/sqrt(H)) * sum_h Q[b,q,h] * K[b,k,h]
// grid = (SEQ/128, SEQ/128, BATCH)
// =====================================================================
__global__ __launch_bounds__(256) void scores_kernel(
    const unsigned short* __restrict__ qb, const unsigned short* __restrict__ kb,
    float* __restrict__ sc) {
  __shared__ unsigned short as[2][128 * BK];
  __shared__ unsigned short bs[2][128 * BK];
  const int b = blockIdx.z;
  const int mBase = blockIdx.x * 128;
  const int nBase = blockIdx.y * 128;
  const int w = threadIdx.x >> 5;
  const int lane = threadIdx.x & 31;
  const unsigned short* Q = qb + (size_t)b * SEQ * HDIM;
  const unsigned short* K = kb + (size_t)b * SEQ * HDIM;

  f32x8 acc[8] = {};
  const int NIT = HDIM / BK;  // 16
  stage_bf16_tile(as[0], Q + (size_t)mBase * HDIM, HDIM);
  stage_bf16_tile(bs[0], K + (size_t)nBase * HDIM, HDIM);
  __syncthreads();
  for (int it = 0; it < NIT; ++it) {
    const int cur = it & 1;
    if (it + 1 < NIT) {
      const int kk = (it + 1) * BK;
      stage_bf16_tile(as[1 ^ cur], Q + (size_t)mBase * HDIM + kk, HDIM);
      stage_bf16_tile(bs[1 ^ cur], K + (size_t)nBase * HDIM + kk, HDIM);
    }
    mma_tile(as[cur], bs[cur], w * 16, acc);
    __syncthreads();
  }

  const float scale = 0.03125f;  // 1/sqrt(1024)
  const int n = lane & 15, h = lane >> 4;
  float* dst = sc + (size_t)b * SEQ * SEQ;
#pragma unroll
  for (int j = 0; j < 8; ++j) {
#pragma unroll
    for (int r = 0; r < 8; ++r) {
      const int mrow = mBase + w * 16 + h * 8 + r;
      const int col = nBase + j * 16 + n;
      dst[(size_t)mrow * SEQ + col] = acc[j][r] * scale;
    }
  }
}

// =====================================================================
// Kernel 3: row softmax over 2048 fp32 (+mask), output bf16 weights.
// grid = (SEQ, BATCH), 256 threads, 8 elements/thread.
// Wave32 shuffle reduction + one 8-entry LDS round across waves.
// =====================================================================
__device__ __forceinline__ float wave_max32(float v) {
#pragma unroll
  for (int off = 16; off > 0; off >>= 1) v = fmaxf(v, __shfl_xor(v, off, 32));
  return v;
}
__device__ __forceinline__ float wave_sum32(float v) {
#pragma unroll
  for (int off = 16; off > 0; off >>= 1) v += __shfl_xor(v, off, 32);
  return v;
}

__global__ __launch_bounds__(256) void softmax_kernel(
    const float* __restrict__ sc, const float* __restrict__ mask,
    unsigned short* __restrict__ wt) {
  __shared__ float red[8];
  const int b = blockIdx.y, row = blockIdx.x, t = threadIdx.x;
  const int w = t >> 5, lane = t & 31;
  const float* src = sc + ((size_t)b * SEQ + row) * SEQ;
  const float* msk = mask + ((size_t)b * SEQ + row) * SEQ;
  unsigned short* dst = wt + ((size_t)b * SEQ + row) * SEQ;

  float v[8];
  float lmax = -3.402823466e38f;
#pragma unroll
  for (int i = 0; i < 8; ++i) {
    const int idx = t + i * 256;
    v[i] = src[idx] + msk[idx];
    lmax = fmaxf(lmax, v[i]);
  }
  lmax = wave_max32(lmax);
  if (lane == 0) red[w] = lmax;
  __syncthreads();
  float rmax = red[0];
#pragma unroll
  for (int i = 1; i < 8; ++i) rmax = fmaxf(rmax, red[i]);
  __syncthreads();  // everyone done reading red before reuse

  float lsum = 0.f;
#pragma unroll
  for (int i = 0; i < 8; ++i) {
    v[i] = __expf(v[i] - rmax);
    lsum += v[i];
  }
  lsum = wave_sum32(lsum);
  if (lane == 0) red[w] = lsum;
  __syncthreads();
  float rsum = red[0];
#pragma unroll
  for (int i = 1; i < 8; ++i) rsum += red[i];
  const float inv = 1.0f / rsum;
#pragma unroll
  for (int i = 0; i < 8; ++i) dst[t + i * 256] = f2bf(v[i] * inv);
}

// =====================================================================
// Kernel 4: out[b,q,h] = sum_k weights[b,q,k] * V[b,k,h]  (V via Vt[H][MTOT])
// grid = (SEQ/128, HDIM/128, BATCH)
// =====================================================================
__global__ __launch_bounds__(256) void pv_kernel(
    const unsigned short* __restrict__ wt, const unsigned short* __restrict__ vtb,
    float* __restrict__ out) {
  __shared__ unsigned short as[2][128 * BK];
  __shared__ unsigned short bs[2][128 * BK];
  const int b = blockIdx.z;
  const int mBase = blockIdx.x * 128;  // query rows
  const int nBase = blockIdx.y * 128;  // output feature columns
  const int w = threadIdx.x >> 5;
  const int lane = threadIdx.x & 31;
  const unsigned short* Wm = wt + (size_t)b * SEQ * SEQ;
  const unsigned short* Vt = vtb + (size_t)b * SEQ;  // column offset into Vt rows

  f32x8 acc[8] = {};
  const int NIT = SEQ / BK;  // 32
  stage_bf16_tile(as[0], Wm + (size_t)mBase * SEQ, SEQ);
  stage_bf16_tile(bs[0], Vt + (size_t)nBase * MTOT, MTOT);
  __syncthreads();
  for (int it = 0; it < NIT; ++it) {
    const int cur = it & 1;
    if (it + 1 < NIT) {
      const int kk = (it + 1) * BK;
      stage_bf16_tile(as[1 ^ cur], Wm + (size_t)mBase * SEQ + kk, SEQ);
      stage_bf16_tile(bs[1 ^ cur], Vt + (size_t)nBase * MTOT + kk, MTOT);
    }
    mma_tile(as[cur], bs[cur], w * 16, acc);
    __syncthreads();
  }

  const int n = lane & 15, h = lane >> 4;
#pragma unroll
  for (int j = 0; j < 8; ++j) {
#pragma unroll
    for (int r = 0; r < 8; ++r) {
      const int mrow = mBase + w * 16 + h * 8 + r;
      const int col = nBase + j * 16 + n;
      out[((size_t)b * SEQ + mrow) * HDIM + col] = acc[j][r];
    }
  }
}

// =====================================================================
extern "C" void kernel_launch(void* const* d_in, const int* in_sizes, int n_in,
                              void* d_out, int out_size, void* d_ws, size_t ws_size,
                              hipStream_t stream) {
  const float* x1 = (const float*)d_in[0];
  const float* x2 = (const float*)d_in[1];
  const float* x3 = (const float*)d_in[2];
  const float* mask = (const float*)d_in[3];
  const float* Wq = (const float*)d_in[4];
  const float* Wk = (const float*)d_in[5];
  const float* Wv = (const float*)d_in[6];
  float* out = (float*)d_out;

  // Workspace layout (all regions fully re-written each call):
  //   [  0MB,  16MB)  Q   bf16 [MTOT][HDIM]
  //   [ 16MB,  32MB)  K   bf16 [MTOT][HDIM]
  //   [ 32MB,  48MB)  Vt  bf16 [HDIM][MTOT]   (transposed V)
  //   [ 48MB, 112MB)  scores fp32 [BATCH][SEQ][SEQ]   (fits in 192MB L2)
  //   [112MB, 144MB)  weights bf16 [BATCH][SEQ][SEQ]
  char* ws = (char*)d_ws;
  unsigned short* qb  = (unsigned short*)(ws);
  unsigned short* kb  = (unsigned short*)(ws + (size_t)16 * 1024 * 1024);
  unsigned short* vtb = (unsigned short*)(ws + (size_t)32 * 1024 * 1024);
  float* sc           = (float*)(ws + (size_t)48 * 1024 * 1024);
  unsigned short* wtb = (unsigned short*)(ws + (size_t)112 * 1024 * 1024);

  dim3 blk(256, 1, 1);
  qkv_proj_kernel<<<dim3(MTOT / 128, HDIM / 128, 3), blk, 0, stream>>>(
      x1, x2, x3, Wq, Wk, Wv, qb, kb, vtb);
  scores_kernel<<<dim3(SEQ / 128, SEQ / 128, BATCH), blk, 0, stream>>>(qb, kb, sc);
  softmax_kernel<<<dim3(SEQ, BATCH, 1), blk, 0, stream>>>(sc, mask, wtb);
  pv_kernel<<<dim3(SEQ / 128, HDIM / 128, BATCH), blk, 0, stream>>>(wtb, vtb, out);
}
